// DeformFlowNet_30090540876298
// MI455X (gfx1250) — matrix operand order repaired and checked by reading, not compiled, not run on hardware
//
#include <hip/hip_runtime.h>
#include <hip/hip_bf16.h>

typedef __attribute__((ext_vector_type(16))) _Float16 v16h;
typedef __attribute__((ext_vector_type(8)))  _Float16 v8h;
typedef __attribute__((ext_vector_type(8)))  float    v8f;
typedef __attribute__((ext_vector_type(4)))  float    v4f;

#define GS_LOOP(i, total) for (long i = (long)blockIdx.x * blockDim.x + threadIdx.x; \
                               i < (total); i += (long)gridDim.x * blockDim.x)

// ---------------------------------------------------------------------------
// Weight pack: f32 (C,K) -> f16 (Cp,Kp), zero padded.
// ---------------------------------------------------------------------------
__global__ void pack_w_kernel(const float* __restrict__ W, _Float16* __restrict__ Wh,
                              long total /*Cp*Kp*/, int C, int K, int Kp) {
    GS_LOOP(i, total) {
        int  k = (int)(i % Kp);
        long c = i / Kp;
        Wh[i] = (c < C && k < K) ? (_Float16)W[c * (size_t)K + k] : (_Float16)0.0f;
    }
}

// ---------------------------------------------------------------------------
// WMMA GEMM:  Y[R,C] = X[R,Kp] @ Wh^T + bias
//   X: f32, row stride Kp (multiple of 32, zero-padded)
//   Wh: f16 packed (Cp,Kp), Cp multiple of 16
//   R: multiple of 16.  Each wave computes a 16 x (NT*16) tile: one A fragment
//   is reused across NT independent WMMA chains (NT=4 for all big layers).
// ---------------------------------------------------------------------------
template <int NT>
__global__ void gemm_wmma_kernel(const float* __restrict__ X, const _Float16* __restrict__ Wh,
                                 const float* __restrict__ bias, float* __restrict__ Y,
                                 int R, int Kp, int C) {
    const int lane    = threadIdx.x;                       // 0..31
    const int rowBase = (blockIdx.y * 4 + threadIdx.y) * 16;
    const int colBase = blockIdx.x * (NT * 16);
    if (rowBase >= R) return;                              // wave-uniform: EXEC stays all-1s
    const int half = lane >> 4;
    const int n    = lane & 15;
    const float*    xrow  = X  + (size_t)(rowBase + n) * Kp;
    const _Float16* wbase = Wh + (size_t)(colBase + n) * Kp;

    v8f acc[NT];
#pragma unroll
    for (int jt = 0; jt < NT; ++jt) acc[jt] = (v8f){};

    for (int kb = 0; kb < Kp; kb += 32) {
        v4f a0 = *(const v4f*)(xrow + kb + half * 8);
        v4f a1 = *(const v4f*)(xrow + kb + half * 8 + 4);
        v4f a2 = *(const v4f*)(xrow + kb + 16 + half * 8);
        v4f a3 = *(const v4f*)(xrow + kb + 16 + half * 8 + 4);
        v16h af;
#pragma unroll
        for (int j = 0; j < 4; ++j) {
            af[j]      = (_Float16)a0[j];
            af[4 + j]  = (_Float16)a1[j];
            af[8 + j]  = (_Float16)a2[j];
            af[12 + j] = (_Float16)a3[j];
        }
#pragma unroll
        for (int jt = 0; jt < NT; ++jt) {
            const _Float16* wrow = wbase + (size_t)jt * 16 * Kp;
            v8h b0 = *(const v8h*)(wrow + kb + half * 8);
            v8h b1 = *(const v8h*)(wrow + kb + 16 + half * 8);
            v16h bf = __builtin_shufflevector(b0, b1, 0, 1, 2, 3, 4, 5, 6, 7,
                                              8, 9, 10, 11, 12, 13, 14, 15);
            acc[jt] = __builtin_amdgcn_wmma_f32_16x16x32_f16(false, af, false, bf,
                                                             (short)0, acc[jt], false, false);
        }
    }

#pragma unroll
    for (int jt = 0; jt < NT; ++jt) {
        int c = colBase + jt * 16 + n;
        float bv = (c < C && bias) ? bias[c] : 0.0f;
#pragma unroll
        for (int v = 0; v < 8; ++v) {
            int row = rowBase + half * 8 + v;              // C/D: M = half*8+v, N = lane&15
            if (c < C) Y[(size_t)row * C + c] = acc[jt][v] + bv;
        }
    }
}

// ---------------------------------------------------------------------------
// BatchNorm: per-channel mean / biased-variance over all rows, then apply.
// ---------------------------------------------------------------------------
__global__ void bn_stats_kernel(const float* __restrict__ X, float* mean, float* var,
                                long R, int C) {
    __shared__ float s1[256], s2[256];
    int c = blockIdx.x;
    float a = 0.0f, b = 0.0f;
    for (long r = threadIdx.x; r < R; r += blockDim.x) {
        float x = X[r * C + c];
        a += x; b += x * x;
    }
    s1[threadIdx.x] = a; s2[threadIdx.x] = b;
    __syncthreads();
    for (int o = 128; o > 0; o >>= 1) {
        if (threadIdx.x < o) { s1[threadIdx.x] += s1[threadIdx.x + o]; s2[threadIdx.x] += s2[threadIdx.x + o]; }
        __syncthreads();
    }
    if (threadIdx.x == 0) {
        float m = s1[0] / (float)R;
        mean[c] = m;
        var[c]  = s2[0] / (float)R - m * m;
    }
}

__global__ void bn_apply_kernel(float* X, const float* __restrict__ mean, const float* __restrict__ var,
                                const float* __restrict__ g, const float* __restrict__ be,
                                long total, int C, int relu) {
    GS_LOOP(i, total) {
        int c = (int)(i % C);
        float x = X[i];
        x = (x - mean[c]) * rsqrtf(var[c] + 1e-5f) * g[c] + be[c];
        if (relu) x = fmaxf(x, 0.0f);
        X[i] = x;
    }
}

// ---------------------------------------------------------------------------
// Farthest point sampling: one block per batch; out[0]=0, first-max tie rule.
// ---------------------------------------------------------------------------
__global__ void fps_kernel(const float* __restrict__ xyz, int* __restrict__ out, int N, int S) {
    __shared__ float dist[4096];
    __shared__ float rv[256];
    __shared__ int   ri[256];
    __shared__ float cent[3];
    const int b = blockIdx.x;
    const int t = threadIdx.x;
    for (int i = t; i < N; i += 256) dist[i] = 1e10f;
    int far = 0;
    for (int s = 0; s < S; ++s) {
        if (t == 0) {
            out[b * S + s] = far;
            cent[0] = xyz[((size_t)b * N + far) * 3 + 0];
            cent[1] = xyz[((size_t)b * N + far) * 3 + 1];
            cent[2] = xyz[((size_t)b * N + far) * 3 + 2];
        }
        __syncthreads();
        float bv = -1.0f; int bi = 0;
        for (int i = t; i < N; i += 256) {
            float dx = xyz[((size_t)b * N + i) * 3 + 0] - cent[0];
            float dy = xyz[((size_t)b * N + i) * 3 + 1] - cent[1];
            float dz = xyz[((size_t)b * N + i) * 3 + 2] - cent[2];
            float d  = dx * dx + dy * dy + dz * dz;
            float nd = fminf(dist[i], d);
            dist[i] = nd;
            if (nd > bv) { bv = nd; bi = i; }
        }
        rv[t] = bv; ri[t] = bi;
        __syncthreads();
        for (int o = 128; o > 0; o >>= 1) {
            if (t < o) {
                if (rv[t + o] > rv[t] || (rv[t + o] == rv[t] && ri[t + o] < ri[t])) {
                    rv[t] = rv[t + o]; ri[t] = ri[t + o];
                }
            }
            __syncthreads();
        }
        far = ri[0];
        __syncthreads();
    }
}

// per (b,s): dst[b,s,:] = src[b, idx[b,s], :]
__global__ void gather3_kernel(const float* __restrict__ src, const int* __restrict__ idx,
                               float* __restrict__ dst, long total /*B*S*/, int N, int S) {
    GS_LOOP(i, total) {
        long b = i / S;
        int j = idx[i];
        dst[i * 3 + 0] = src[((size_t)b * N + j) * 3 + 0];
        dst[i * 3 + 1] = src[((size_t)b * N + j) * 3 + 1];
        dst[i * 3 + 2] = src[((size_t)b * N + j) * 3 + 2];
    }
}

// Ball query: first K indices (ascending) with d2 <= r2, padded with first hit.
__global__ void ball_query_kernel(const float* __restrict__ xyz, const float* __restrict__ new_xyz,
                                  int* __restrict__ idx, long total /*B*S*/, int N, int S,
                                  int K, float r2) {
    GS_LOOP(i, total) {
        long b = i / S;
        float qx = new_xyz[i * 3 + 0], qy = new_xyz[i * 3 + 1], qz = new_xyz[i * 3 + 2];
        int cnt = 0, first = 0; bool found = false;
        for (int j = 0; j < N && cnt < K; ++j) {
            float dx = xyz[((size_t)b * N + j) * 3 + 0] - qx;
            float dy = xyz[((size_t)b * N + j) * 3 + 1] - qy;
            float dz = xyz[((size_t)b * N + j) * 3 + 2] - qz;
            float d2 = dx * dx + dy * dy + dz * dz;
            if (d2 <= r2) {
                if (!found) { first = j; found = true; }
                idx[i * K + cnt++] = j;
            }
        }
        for (; cnt < K; ++cnt) idx[i * K + cnt] = first;
    }
}

// Grouped features, padded: out rows = B*S*K, width Cd; cols [3+D,Cd) zeroed.
__global__ void group_concat_kernel(const float* __restrict__ xyz, const float* __restrict__ new_xyz,
                                    const float* __restrict__ pts, const int* __restrict__ idx,
                                    float* __restrict__ out, long total, int N, int S, int K,
                                    int D, int Cd) {
    GS_LOOP(i, total) {
        int c = (int)(i % Cd);
        long r = i / Cd;
        int k = (int)(r % K);
        long bs = r / K;
        long b = bs / S;
        float v = 0.0f;
        if (c < 3 + D) {
            int j = idx[bs * K + k];
            if (c < 3) v = xyz[((size_t)b * N + j) * 3 + c] - new_xyz[bs * 3 + c];
            else       v = pts[((size_t)b * N + j) * D + (c - 3)];
        }
        out[i] = v;
    }
}

// max over group axis: X (B*S*K, C) -> Y (B*S, C)
__global__ void max_group_kernel(const float* __restrict__ X, float* __restrict__ Y,
                                 long total /*B*S*C*/, int K, int C) {
    GS_LOOP(i, total) {
        int c = (int)(i % C);
        long rs = i / C;
        float m = -3.4e38f;
        for (int k = 0; k < K; ++k) m = fmaxf(m, X[((size_t)rs * K + k) * C + c]);
        Y[i] = m;
    }
}

// max over axis 1: X (B,N,C) -> Y (B,C)
__global__ void max_axis1_kernel(const float* __restrict__ X, float* __restrict__ Y,
                                 long total /*B*C*/, int N, int C) {
    GS_LOOP(i, total) {
        int c = (int)(i % C);
        long b = i / C;
        float m = -3.4e38f;
        for (int n = 0; n < N; ++n) m = fmaxf(m, X[((size_t)b * N + n) * C + c]);
        Y[i] = m;
    }
}

// (B,C,M) -> (B*M, C)   (unpadded, used for geometry width 3)
__global__ void bcm_to_bmc_kernel(const float* __restrict__ X, float* __restrict__ Y,
                                  long total, int C, int M) {
    GS_LOOP(i, total) {
        int c = (int)(i % C);
        long r = i / C;
        long b = r / M;
        long m = r % M;
        Y[i] = X[(size_t)b * C * M + (size_t)c * M + m];
    }
}

// (B,C,M) -> (B*M, Cd) with zero pad cols [C,Cd)
__global__ void bcm_to_bmc_pad_kernel(const float* __restrict__ X, float* __restrict__ Y,
                                      long total /*B*M*Cd*/, int C, int M, int Cd) {
    GS_LOOP(i, total) {
        int c = (int)(i % Cd);
        long r = i / Cd;
        long b = r / M;
        long m = r % M;
        Y[i] = (c < C) ? X[(size_t)b * C * M + (size_t)c * M + m] : 0.0f;
    }
}

// (B*M, C) -> (B,C,M)
__global__ void bmc_to_bcm_kernel(const float* __restrict__ X, float* __restrict__ Y,
                                  long total, int C, int M) {
    GS_LOOP(i, total) {
        long m = i % M;
        long c = (i / M) % C;
        long b = i / ((long)C * M);
        Y[i] = X[((size_t)b * M + m) * C + c];
    }
}

// 3 nearest neighbors + normalized inverse-distance weights.
__global__ void knn3_kernel(const float* __restrict__ xyz1, const float* __restrict__ xyz2,
                            int* __restrict__ idx3, float* __restrict__ w3,
                            long total /*B*N1*/, int N1, int N2) {
    GS_LOOP(i, total) {
        long b = i / N1;
        float px = xyz1[i * 3 + 0], py = xyz1[i * 3 + 1], pz = xyz1[i * 3 + 2];
        float d0 = 1e30f, d1 = 1e30f, d2v = 1e30f;
        int i0 = 0, i1 = 0, i2 = 0;
        for (int j = 0; j < N2; ++j) {
            float dx = px - xyz2[((size_t)b * N2 + j) * 3 + 0];
            float dy = py - xyz2[((size_t)b * N2 + j) * 3 + 1];
            float dz = pz - xyz2[((size_t)b * N2 + j) * 3 + 2];
            float d = dx * dx + dy * dy + dz * dz;
            if (d < d0)      { d2v = d1; i2 = i1; d1 = d0; i1 = i0; d0 = d; i0 = j; }
            else if (d < d1) { d2v = d1; i2 = i1; d1 = d;  i1 = j; }
            else if (d < d2v){ d2v = d;  i2 = j; }
        }
        float w0 = 1.0f / (d0 + 1e-8f), w1 = 1.0f / (d1 + 1e-8f), w2 = 1.0f / (d2v + 1e-8f);
        float s = w0 + w1 + w2;
        idx3[i * 3 + 0] = i0; idx3[i * 3 + 1] = i1; idx3[i * 3 + 2] = i2;
        w3[i * 3 + 0] = w0 / s; w3[i * 3 + 1] = w1 / s; w3[i * 3 + 2] = w2 / s;
    }
}

// dst[r*Cd + off + c] = src[r*Cs + c]
__global__ void copy_cols_kernel(const float* __restrict__ src, float* __restrict__ dst,
                                 long total /*R*Cs*/, int Cs, int Cd, int off) {
    GS_LOOP(i, total) {
        int c = (int)(i % Cs);
        long r = i / Cs;
        dst[r * (size_t)Cd + off + c] = src[i];
    }
}

// zero cols [from,Cd) of dst (rows x Cd)
__global__ void zero_cols_kernel(float* __restrict__ dst, long total /*rows*(Cd-from)*/,
                                 int Cd, int from) {
    const int w = Cd - from;
    GS_LOOP(i, total) {
        int c = (int)(i % w);
        long r = i / w;
        dst[r * (size_t)Cd + from + c] = 0.0f;
    }
}

// broadcast (B,Cs) into (B*N1, Cd) at column offset
__global__ void bcast_cols_kernel(const float* __restrict__ src, float* __restrict__ dst,
                                  long total /*B*N1*Cs*/, int N1, int Cs, int Cd, int off) {
    GS_LOOP(i, total) {
        int c = (int)(i % Cs);
        long r = i / Cs;
        long b = r / N1;
        dst[r * (size_t)Cd + off + c] = src[b * Cs + c];
    }
}

// interpolate points2 (B,N2,C2) at 3-NN -> dst cols at offset
__global__ void interp_cols_kernel(const float* __restrict__ pts2, const int* __restrict__ idx3,
                                   const float* __restrict__ w3, float* __restrict__ dst,
                                   long total /*B*N1*C2*/, int N1, int N2, int C2, int Cd, int off) {
    GS_LOOP(i, total) {
        int c = (int)(i % C2);
        long r = i / C2;    // b*N1 + n1
        long b = r / N1;
        float acc = 0.0f;
        for (int j = 0; j < 3; ++j) {
            int id = idx3[r * 3 + j];
            acc += pts2[((size_t)b * N2 + id) * C2 + c] * w3[r * 3 + j];
        }
        dst[r * (size_t)Cd + off + c] = acc;
    }
}

// y[p,0:3] = x[p,0:3] @ (trans(b)+I3); x,y row stride 32; pad cols zeroed.
__global__ void apply_trans_pad_kernel(const float* __restrict__ x, const float* __restrict__ trans9,
                                       float* __restrict__ y, long total /*B*NP*/, int NP) {
    GS_LOOP(i, total) {
        long b = i / NP;
        float x0 = x[i * 32 + 0], x1 = x[i * 32 + 1], x2 = x[i * 32 + 2];
        const float* t = &trans9[b * 9];
#pragma unroll
        for (int j = 0; j < 3; ++j) {
            float v = x0 * (t[0 * 3 + j] + (j == 0 ? 1.0f : 0.0f))
                    + x1 * (t[1 * 3 + j] + (j == 1 ? 1.0f : 0.0f))
                    + x2 * (t[2 * 3 + j] + (j == 2 ? 1.0f : 0.0f));
            y[i * 32 + j] = v;
        }
        for (int j = 3; j < 32; ++j) y[i * 32 + j] = 0.0f;
    }
}

__global__ void copy_flat_kernel(const float* __restrict__ src, float* __restrict__ dst, int n) {
    int i = blockIdx.x * blockDim.x + threadIdx.x;
    if (i < n) dst[i] = src[i];
}

// ---------------------------------------------------------------------------
// Host orchestration
// ---------------------------------------------------------------------------
namespace {
struct Ly { const float *W, *b, *g, *be; _Float16* Wh; int C, K, Kp, Cp; };
}

extern "C" void kernel_launch(void* const* d_in, const int* in_sizes, int n_in,
                              void* d_out, int out_size, void* d_ws, size_t ws_size,
                              hipStream_t stream) {
    (void)in_sizes; (void)n_in; (void)out_size; (void)ws_size;
    const int  Bn = 16, M = 4096, NPp = 2048;
    const float* xyz_template = (const float*)d_in[0];
    const float* xyz_partial  = (const float*)d_in[1];
    float* out = (float*)d_out;

    // ---- workspace bump allocator ----
    float* F = (float*)d_ws;
    size_t off = 0;
    auto allocF = [&](size_t n) { float* p = F + off; off += (n + 255) & ~(size_t)255; return p; };
    float* SA   = allocF((size_t)34 << 20);   // ping
    float* SB   = allocF((size_t)68 << 20);   // pong (holds sa3's 65536x1024)
    float* l0xyz = allocF((size_t)Bn * M * 3);
    float* pxp   = allocF((size_t)Bn * NPp * 32);   // padded (B*NP,32), cols 0..2 = partial xyz
    float* px2p  = allocF((size_t)Bn * NPp * 32);   // after STN transform, padded
    float* l1xyz = allocF((size_t)Bn * 512 * 3);
    float* l2xyz = allocF((size_t)Bn * 128 * 3);
    float* l3xyz = allocF((size_t)Bn * 32 * 3);
    float* l1p   = allocF((size_t)Bn * 512 * 128);
    float* l2p   = allocF((size_t)Bn * 128 * 256);
    float* l3p   = allocF((size_t)Bn * 32 * 1024);
    float* l2pn  = allocF((size_t)Bn * 128 * 256);
    float* l1pn  = allocF((size_t)Bn * 512 * 128);
    float* pfeat = allocF((size_t)Bn * 1024);
    float* l3max = allocF((size_t)Bn * 1024);
    float* hb    = allocF((size_t)Bn * 1024);
    float* h1    = allocF((size_t)Bn * 512);
    float* h2    = allocF((size_t)Bn * 256);
    float* trans = allocF((size_t)Bn * 9);
    float* U     = allocF((size_t)Bn * 2048);
    float* u1    = allocF((size_t)Bn * 128);
    float* u2    = allocF((size_t)Bn * 2);
    float* mean  = allocF(1024);
    float* var   = allocF(1024);
    float* k3w   = allocF((size_t)Bn * 128 * 3);
    float* k2w   = allocF((size_t)Bn * 512 * 3);
    float* k1w   = allocF((size_t)Bn * 4096 * 3);
    int* fps1  = (int*)allocF((size_t)Bn * 512);
    int* fps2  = (int*)allocF((size_t)Bn * 128);
    int* fps3  = (int*)allocF((size_t)Bn * 32);
    int* bidx1 = (int*)allocF((size_t)Bn * 512 * 32);
    int* bidx2 = (int*)allocF((size_t)Bn * 128 * 64);
    int* bidx3 = (int*)allocF((size_t)Bn * 32 * 128);
    int* k3i   = (int*)allocF((size_t)Bn * 128 * 3);
    int* k2i   = (int*)allocF((size_t)Bn * 512 * 3);
    int* k1i   = (int*)allocF((size_t)Bn * 4096 * 3);
    _Float16* Wark = (_Float16*)allocF((size_t)4 << 20);  // f16 weight arena (16 MB)
    size_t woff = 0;

    auto nb = [](long t) { long b = (t + 255) / 256; if (b < 1) b = 1; if (b > 32768) b = 32768; return (int)b; };

    auto P  = [&](int i) { return (const float*)d_in[i]; };
    // params flatten order (jax pytree, keys sorted; per-layer W,b,be,g):
    auto mk = [&](int i, int C, int K, bool bn) {
        Ly ly;
        ly.W = P(i); ly.b = P(i + 1);
        ly.be = bn ? P(i + 2) : nullptr;
        ly.g  = bn ? P(i + 3) : nullptr;
        ly.C = C; ly.K = K;
        ly.Kp = (K + 31) & ~31;
        ly.Cp = (C + 15) & ~15;
        ly.Wh = Wark + woff;
        woff += (size_t)ly.Cp * ly.Kp;
        long tot = (long)ly.Cp * ly.Kp;
        pack_w_kernel<<<nb(tot), 256, 0, stream>>>(ly.W, ly.Wh, tot, C, K, ly.Kp);
        return ly;
    };
    Ly conv1 = mk(2, 128, 128, true),   conv2 = mk(6, 3, 128, false);
    Ly cu1   = mk(8, 128, 2048, true),  cu2   = mk(12, 2, 128, false);
    Ly enc0  = mk(14, 64, 3, true),     enc1  = mk(18, 128, 64, true),  enc2 = mk(22, 1024, 128, true);
    Ly fp1_0 = mk(26, 128, 131, true),  fp1_1 = mk(30, 128, 128, true), fp1_2 = mk(34, 128, 128, true);
    Ly fp2_0 = mk(38, 256, 384, true),  fp2_1 = mk(42, 128, 256, true);
    Ly fp3_0 = mk(46, 256, 2307, true), fp3_1 = mk(50, 256, 256, true);
    Ly sa1_0 = mk(54, 64, 6, true),     sa1_1 = mk(58, 64, 64, true),   sa1_2 = mk(62, 128, 64, true);
    Ly sa2_0 = mk(66, 128, 131, true),  sa2_1 = mk(70, 128, 128, true), sa2_2 = mk(74, 256, 128, true);
    Ly sa3_0 = mk(78, 256, 259, true),  sa3_1 = mk(82, 512, 256, true), sa3_2 = mk(86, 1024, 512, true);
    Ly stn0  = mk(90, 64, 3, true),     stn1  = mk(94, 128, 64, true),  stn2 = mk(98, 1024, 128, true);
    Ly fc1   = mk(102, 512, 1024, true),fc2   = mk(106, 256, 512, true),fc3  = mk(110, 9, 256, false);

    auto gemm = [&](const float* X, const Ly& ly, float* Y, long R) {
        dim3 blk(32, 4);
        unsigned gy = (unsigned)((R + 63) / 64);
        if (ly.Cp == 16) {
            dim3 grd(1, gy);
            gemm_wmma_kernel<1><<<grd, blk, 0, stream>>>(X, ly.Wh, ly.b, Y, (int)R, ly.Kp, ly.C);
        } else {          // all other Cp are multiples of 64
            dim3 grd(ly.Cp / 64, gy);
            gemm_wmma_kernel<4><<<grd, blk, 0, stream>>>(X, ly.Wh, ly.b, Y, (int)R, ly.Kp, ly.C);
        }
    };
    auto dbr = [&](const float* X, const Ly& ly, float* Y, long R, int relu) {
        gemm(X, ly, Y, R);
        bn_stats_kernel<<<ly.C, 256, 0, stream>>>(Y, mean, var, R, ly.C);
        long tot = R * ly.C;
        bn_apply_kernel<<<nb(tot), 256, 0, stream>>>(Y, mean, var, ly.g, ly.be, tot, ly.C, relu);
    };

    // ---- l0: transpose template (B,3,M) -> (B*M,3); l0_points == l0_xyz ----
    long t = (long)Bn * M * 3;
    bcm_to_bmc_kernel<<<nb(t), 256, 0, stream>>>(xyz_template, l0xyz, t, 3, M);

    // ---- set abstraction stages ----
    auto sa_stage = [&](const float* xyz_in, int N, const float* pts_in, int D, int S, int K,
                        float radius, const Ly& a, const Ly& b, const Ly& c,
                        int* fpsbuf, int* ballbuf, float* new_xyz, float* out_pts) {
        fps_kernel<<<Bn, 256, 0, stream>>>(xyz_in, fpsbuf, N, S);
        long tS = (long)Bn * S;
        gather3_kernel<<<nb(tS), 256, 0, stream>>>(xyz_in, fpsbuf, new_xyz, tS, N, S);
        ball_query_kernel<<<nb(tS), 256, 0, stream>>>(xyz_in, new_xyz, ballbuf, tS, N, S, K, radius * radius);
        long rows = (long)Bn * S * K;
        long tot = rows * a.Kp;
        group_concat_kernel<<<nb(tot), 256, 0, stream>>>(xyz_in, new_xyz, pts_in, ballbuf, SA, tot,
                                                         N, S, K, D, a.Kp);
        dbr(SA, a, SB, rows, 1);
        dbr(SB, b, SA, rows, 1);
        dbr(SA, c, SB, rows, 1);
        long tOut = (long)Bn * S * c.C;
        max_group_kernel<<<nb(tOut), 256, 0, stream>>>(SB, out_pts, tOut, K, c.C);
    };
    sa_stage(l0xyz, M,   l0xyz, 3,   512, 32,  0.2f, sa1_0, sa1_1, sa1_2, fps1, bidx1, l1xyz, l1p);
    sa_stage(l1xyz, 512, l1p,   128, 128, 64,  0.4f, sa2_0, sa2_1, sa2_2, fps2, bidx2, l2xyz, l2p);
    sa_stage(l2xyz, 128, l2p,   256, 32,  128, 0.8f, sa3_0, sa3_1, sa3_2, fps3, bidx3, l3xyz, l3p);

    // ---- pointnet encoder on partial cloud ----
    long rp = (long)Bn * NPp;
    bcm_to_bmc_pad_kernel<<<nb(rp * 32), 256, 0, stream>>>(xyz_partial, pxp, rp * 32, 3, NPp, 32);
    dbr(pxp, stn0, SA, rp, 1);
    dbr(SA, stn1, SB, rp, 1);
    dbr(SB, stn2, SA, rp, 1);
    long tBC = (long)Bn * 1024;
    max_axis1_kernel<<<nb(tBC), 256, 0, stream>>>(SA, hb, tBC, NPp, 1024);
    dbr(hb, fc1, h1, Bn, 1);
    dbr(h1, fc2, h2, Bn, 1);
    gemm(h2, fc3, trans, Bn);
    apply_trans_pad_kernel<<<nb(rp), 256, 0, stream>>>(pxp, trans, px2p, rp, NPp);
    dbr(px2p, enc0, SA, rp, 1);
    dbr(SA,   enc1, SB, rp, 1);
    dbr(SB,   enc2, SA, rp, 0);   // last enc conv: bn, no relu
    max_axis1_kernel<<<nb(tBC), 256, 0, stream>>>(SA, pfeat, tBC, NPp, 1024);

    // ---- fp3: l2 <- l3 ----  (concat width 2307 padded to 2336)
    long t128 = (long)Bn * 128;
    knn3_kernel<<<nb(t128), 256, 0, stream>>>(l2xyz, l3xyz, k3i, k3w, t128, 128, 32);
    bcast_cols_kernel<<<nb(t128 * 1024), 256, 0, stream>>>(pfeat, SA, t128 * 1024, 128, 1024, 2336, 0);
    copy_cols_kernel<<<nb(t128 * 3), 256, 0, stream>>>(l2xyz, SA, t128 * 3, 3, 2336, 1024);
    copy_cols_kernel<<<nb(t128 * 256), 256, 0, stream>>>(l2p, SA, t128 * 256, 256, 2336, 1027);
    interp_cols_kernel<<<nb(t128 * 1024), 256, 0, stream>>>(l3p, k3i, k3w, SA, t128 * 1024, 128, 32, 1024, 2336, 1283);
    zero_cols_kernel<<<nb(t128 * 29), 256, 0, stream>>>(SA, t128 * 29, 2336, 2307);
    dbr(SA, fp3_0, SB, t128, 1);
    dbr(SB, fp3_1, l2pn, t128, 1);

    // ---- fp2: l1 <- l2 ----  (width 384, already aligned)
    long t512 = (long)Bn * 512;
    knn3_kernel<<<nb(t512), 256, 0, stream>>>(l1xyz, l2xyz, k2i, k2w, t512, 512, 128);
    copy_cols_kernel<<<nb(t512 * 128), 256, 0, stream>>>(l1p, SA, t512 * 128, 128, 384, 0);
    interp_cols_kernel<<<nb(t512 * 256), 256, 0, stream>>>(l2pn, k2i, k2w, SA, t512 * 256, 512, 128, 256, 384, 128);
    dbr(SA, fp2_0, SB, t512, 1);
    dbr(SB, fp2_1, l1pn, t512, 1);

    // ---- fp1: l0 <- l1 ----  (concat width 131 padded to 160)
    long t4096 = (long)Bn * 4096;
    knn3_kernel<<<nb(t4096), 256, 0, stream>>>(l0xyz, l1xyz, k1i, k1w, t4096, 4096, 512);
    copy_cols_kernel<<<nb(t4096 * 3), 256, 0, stream>>>(l0xyz, SA, t4096 * 3, 3, 160, 0);
    interp_cols_kernel<<<nb(t4096 * 128), 256, 0, stream>>>(l1pn, k1i, k1w, SA, t4096 * 128, 4096, 512, 128, 160, 3);
    zero_cols_kernel<<<nb(t4096 * 29), 256, 0, stream>>>(SA, t4096 * 29, 160, 131);
    dbr(SA, fp1_0, SB, t4096, 1);
    dbr(SB, fp1_1, SA, t4096, 1);
    dbr(SA, fp1_2, SB, t4096, 1);

    // ---- heads: flow ----
    dbr(SB, conv1, SA, t4096, 1);
    gemm(SA, conv2, SB, t4096);
    t = (long)Bn * 3 * M;
    bmc_to_bcm_kernel<<<nb(t), 256, 0, stream>>>(SB, out, t, 3, M);

    // ---- heads: uncertainty ----
    max_axis1_kernel<<<nb(tBC), 256, 0, stream>>>(l3p, l3max, tBC, 32, 1024);
    copy_cols_kernel<<<nb(tBC), 256, 0, stream>>>(pfeat, U, tBC, 1024, 2048, 0);
    copy_cols_kernel<<<nb(tBC), 256, 0, stream>>>(l3max, U, tBC, 1024, 2048, 1024);
    dbr(U, cu1, u1, Bn, 1);
    gemm(u1, cu2, u2, Bn);
    copy_flat_kernel<<<1, 32, 0, stream>>>(u2, out + (size_t)Bn * 3 * M, Bn * 2);
}
